// GCN_31301721653775
// MI455X (gfx1250) — matrix operand order repaired and checked by reading, not compiled, not run on hardware
//
#include <hip/hip_runtime.h>
#include <hip/hip_bf16.h>

typedef __attribute__((ext_vector_type(16))) __bf16 v16bf;
typedef __attribute__((ext_vector_type(8)))  float  v8f;

#define NF 96          // features == hidden
#define WAVES_PER_BLK 8
#define KT 3           // K chunks of 32 (96 = 3*32)
#define NT 6           // column tiles of 16 (96 = 6*16)

// ---------------------------------------------------------------------------
// Degree kernels: deg starts at 1 (self loop), accumulate in-degree over dst,
// then dinv = rsqrt(deg).
// ---------------------------------------------------------------------------
__global__ __launch_bounds__(256) void gcn_deg_init(float* __restrict__ deg, int n) {
    int i = blockIdx.x * 256 + threadIdx.x;
    if (i < n) deg[i] = 1.0f;
}

__global__ __launch_bounds__(256) void gcn_deg_accum(const long long* __restrict__ dst,
                                                     float* __restrict__ deg, long long e) {
    long long i = (long long)blockIdx.x * 256 + threadIdx.x;
    if (i < e) unsafeAtomicAdd(&deg[dst[i]], 1.0f);
}

__global__ __launch_bounds__(256) void gcn_deg_finish(float* __restrict__ deg, int n) {
    int i = blockIdx.x * 256 + threadIdx.x;
    if (i < n) deg[i] = __frsqrt_rn(deg[i]);
}

// ---------------------------------------------------------------------------
// WMMA GEMM: H[N,96] = relu?(X)[N,96] @ W[96,96]  (bf16 in, fp32 accumulate)
// One wave -> one 16-row tile, all 6 column tiles. K = 3 chunks of 32.
//
// W is staged into LDS pre-permuted into WMMA B-fragment order:
//   tile t = kc*NT + nt; lane l holds 16 contiguous bf16 (32 B, 32 B-aligned)
//   element j of lane l:  k = kc*32 + (j&7) + 8*(l>=16) + 16*(j>>3),  n = nt*16 + (l&15)
// so the per-WMMA B operand load is a single 32 B LDS read (ds_load_b128 x2).
//
// A-fragment (16x32 bf16, ISA 7.12.2) per lane is two contiguous 8-float runs:
//   elems 0..7  <- xrow[kbase + 8*half + 0..7]
//   elems 8..15 <- xrow[kbase + 16 + 8*half + 0..7]
// both 16 B-aligned -> global_load_b128 pairs, then cvt to bf16.
// ---------------------------------------------------------------------------
__global__ __launch_bounds__(256)
void gcn_gemm_wmma(const float* __restrict__ X, const float* __restrict__ W,
                   float* __restrict__ H, int n_rowtiles, int n_rows, int relu_in) {
    __shared__ __bf16 Wlds[KT * NT * 32 * 16];   // 18432 B, B-fragment order

    for (int idx = threadIdx.x; idx < KT * NT * 32 * 16; idx += 256) {
        const int tile = idx >> 9;        // / (32*16)
        const int rem  = idx & 511;
        const int l    = rem >> 4;
        const int j    = rem & 15;
        const int kc   = tile / NT;
        const int nt   = tile % NT;
        const int k    = kc * 32 + (j & 7) + ((l >= 16) ? 8 : 0) + ((j >> 3) << 4);
        const int n    = nt * 16 + (l & 15);
        Wlds[idx] = (__bf16)W[k * NF + n];
    }
    __syncthreads();

    const int wave = threadIdx.x >> 5;
    const int lane = threadIdx.x & 31;
    const int rt   = blockIdx.x * WAVES_PER_BLK + wave;
    if (rt >= n_rowtiles) return;          // wave-uniform: EXEC stays all-1s for WMMA

    const int half = lane >> 4;            // 0: lanes 0-15, 1: lanes 16-31
    const int mrow = lane & 15;

    v8f acc[NT] = {};                      // 6 col tiles x 8 VGPRs f32

    int arow = rt * 16 + mrow;
    if (arow >= n_rows) arow = n_rows - 1; // clamp (keeps EXEC full)
    const float* __restrict__ xrow = X + (size_t)arow * NF;

    for (int kc = 0; kc < KT; ++kc) {
        const int kbase = kc * 32;
        // ---- A fragment: two contiguous 8-float runs -> v16bf ----
        float r0[8], r1[8];
#pragma unroll
        for (int j = 0; j < 8; ++j) {
            r0[j] = xrow[kbase + 8 * half + j];
            r1[j] = xrow[kbase + 16 + 8 * half + j];
        }
        if (relu_in) {
#pragma unroll
            for (int j = 0; j < 8; ++j) {
                r0[j] = fmaxf(r0[j], 0.0f);
                r1[j] = fmaxf(r1[j], 0.0f);
            }
        }
        v16bf a;
#pragma unroll
        for (int j = 0; j < 8; ++j) {
            a[j]     = (__bf16)r0[j];
            a[j + 8] = (__bf16)r1[j];
        }

#pragma unroll
        for (int nt = 0; nt < NT; ++nt) {
            const v16bf b =
                *(const v16bf*)&Wlds[(((kc * NT + nt) * 32) + lane) << 4];
            acc[nt] = __builtin_amdgcn_wmma_f32_16x16x32_bf16(
                false, a, false, b, (short)0, acc[nt], false, false);
        }
    }

    // ---- store D: VGPR d -> M = d + 8*half, N = lane&15 ----
#pragma unroll
    for (int nt = 0; nt < NT; ++nt) {
        const int col = nt * 16 + mrow;
#pragma unroll
        for (int d = 0; d < 8; ++d) {
            int row = rt * 16 + d + (half ? 8 : 0);
            if (row < n_rows) H[(size_t)row * NF + col] = acc[nt][d];
        }
    }
}

// ---------------------------------------------------------------------------
// agg[v,f] = h[v,f] * dinv[v]^2 + b[f]   (self-loop term + bias)
// ---------------------------------------------------------------------------
__global__ __launch_bounds__(256)
void gcn_agg_init(const float* __restrict__ H, const float* __restrict__ dinv,
                  const float* __restrict__ bias, float* __restrict__ out, int n) {
    long long i = (long long)blockIdx.x * 256 + threadIdx.x;
    if (i < (long long)n * NF) {
        int v = (int)(i / NF);
        int f = (int)(i % NF);
        float di = dinv[v];
        out[i] = H[i] * di * di + bias[f];
    }
}

// ---------------------------------------------------------------------------
// Edge scatter: one wave per edge, 3 features per lane (96 = 3*32, wave32).
// agg[dst,f] += h[src,f] * dinv[src]*dinv[dst]   via global_atomic_add_f32.
// h/agg are L2-resident (19.2 MB << 192 MB) -> atomics resolve in L2.
// ---------------------------------------------------------------------------
__global__ __launch_bounds__(256)
void gcn_scatter(const float* __restrict__ H, const long long* __restrict__ src,
                 const long long* __restrict__ dst, const float* __restrict__ dinv,
                 float* __restrict__ out, long long n_edges) {
    long long w = ((long long)blockIdx.x * 256 + threadIdx.x) >> 5;
    int lane = threadIdx.x & 31;
    if (w >= n_edges) return;
    long long s = src[w];
    long long d = dst[w];
    float nrm = dinv[s] * dinv[d];
    const float* __restrict__ hs = H + s * NF;
    float* __restrict__ od = out + d * NF;
#pragma unroll
    for (int i = 0; i < 3; ++i) {
        int f = lane + i * 32;
        unsafeAtomicAdd(&od[f], hs[f] * nrm);
    }
}

// ---------------------------------------------------------------------------
extern "C" void kernel_launch(void* const* d_in, const int* in_sizes, int n_in,
                              void* d_out, int out_size, void* d_ws, size_t ws_size,
                              hipStream_t stream) {
    const float* x  = (const float*)d_in[0];
    const float* W1 = (const float*)d_in[1];
    const float* b1 = (const float*)d_in[2];
    const float* W2 = (const float*)d_in[3];
    const float* b2 = (const float*)d_in[4];
    const float* W3 = (const float*)d_in[5];
    const float* b3 = (const float*)d_in[6];
    const long long* eidx = (const long long*)d_in[7];

    const int       N = in_sizes[0] / NF;
    const long long E = (long long)in_sizes[7] / 2;
    const long long* esrc = eidx;
    const long long* edst = eidx + E;

    // workspace layout: dinv[N] | h[N*96] | A[N*96]
    char* ws = (char*)d_ws;
    float* dinv = (float*)ws;
    size_t off = (((size_t)N * 4) + 255) & ~(size_t)255;
    float* h = (float*)(ws + off);
    float* A = (float*)(ws + off + (size_t)N * NF * sizeof(float));
    float* out = (float*)d_out;

    const int n_rt = (N + 15) / 16;
    const int gemm_blocks = (n_rt + WAVES_PER_BLK - 1) / WAVES_PER_BLK;
    const int nf_blocks   = (int)(((long long)N * NF + 255) / 256);
    const int edge_blocks = (int)((E * 32 + 255) / 256);
    const int n_blocks    = (N + 255) / 256;
    const int e_blocks    = (int)((E + 255) / 256);

    // degrees with self-loops -> dinv
    gcn_deg_init  <<<n_blocks, 256, 0, stream>>>(dinv, N);
    gcn_deg_accum <<<e_blocks, 256, 0, stream>>>(edst, dinv, E);
    gcn_deg_finish<<<n_blocks, 256, 0, stream>>>(dinv, N);

    // ---- layer 1: x -> A ----
    gcn_gemm_wmma<<<gemm_blocks, 256, 0, stream>>>(x, W1, h, n_rt, N, 0);
    gcn_agg_init <<<nf_blocks, 256, 0, stream>>>(h, dinv, b1, A, N);
    gcn_scatter  <<<edge_blocks, 256, 0, stream>>>(h, esrc, edst, dinv, A, E);

    // ---- layer 2: relu(A) -> A  (ReLU fused into GEMM load) ----
    gcn_gemm_wmma<<<gemm_blocks, 256, 0, stream>>>(A, W2, h, n_rt, N, 1);
    gcn_agg_init <<<nf_blocks, 256, 0, stream>>>(h, dinv, b2, A, N);
    gcn_scatter  <<<edge_blocks, 256, 0, stream>>>(h, esrc, edst, dinv, A, E);

    // ---- layer 3: relu(A) -> d_out ----
    gcn_gemm_wmma<<<gemm_blocks, 256, 0, stream>>>(A, W3, h, n_rt, N, 1);
    gcn_agg_init <<<nf_blocks, 256, 0, stream>>>(h, dinv, b3, out, N);
    gcn_scatter  <<<edge_blocks, 256, 0, stream>>>(h, esrc, edst, dinv, out, E);
}
